// BERTSelfAttention_50130858279252
// MI455X (gfx1250) — matrix operand order repaired and checked by reading, not compiled, not run on hardware
//
#include <hip/hip_runtime.h>

// Problem constants (BERT-base shapes from the reference)
#define Bv  8
#define Sv  2048
#define Hv  768
#define NHv 12
#define HDv 64

typedef _Float16 v16h __attribute__((ext_vector_type(16)));
typedef _Float16 v8h  __attribute__((ext_vector_type(8)));
typedef _Float16 v4h  __attribute__((ext_vector_type(4)));
typedef float    v8f  __attribute__((ext_vector_type(8)));
typedef int      b128i __attribute__((vector_size(16)));  // builtin's b128 unit

#if __has_builtin(__builtin_amdgcn_global_load_async_to_lds_b128) && \
    __has_builtin(__builtin_amdgcn_s_wait_asynccnt)
#define USE_ASYNC_LDS 1
#define GPTR(p) ((__attribute__((address_space(1))) b128i*)(p))
#define LPTR(p) ((__attribute__((address_space(3))) b128i*)(p))
#else
#define USE_ASYNC_LDS 0
#endif

__device__ __forceinline__ v16h ldA(const _Float16* lo, const _Float16* hi) {
  v8h a = *(const v8h*)lo;
  v8h b = *(const v8h*)hi;
  v16h r;
#pragma unroll
  for (int i = 0; i < 8; ++i) { r[i] = a[i]; r[i + 8] = b[i]; }
  return r;
}

__device__ __forceinline__ v8f wmma_f16(v16h a, v16h b, v8f c) {
  return __builtin_amdgcn_wmma_f32_16x16x32_f16(false, a, false, b, (short)0, c,
                                                false, false);
}

// ---------------------------------------------------------------------------
// Kernel 0: one-shot fp32 -> f16 conversion (x and the three W matrices).
// After this, x+W in f16 (~28 MB) fit in the 192 MB L2 for all GEMM re-reads.
// ---------------------------------------------------------------------------
__global__ __launch_bounds__(256) void cvt_f32_to_f16(
    const float* __restrict__ src, _Float16* __restrict__ dst, int n4) {
  const int i = blockIdx.x * 256 + threadIdx.x;
  if (i < n4) {
    const float4 f = ((const float4*)src)[i];
    v4h h;
    h[0] = (_Float16)f.x; h[1] = (_Float16)f.y;
    h[2] = (_Float16)f.z; h[3] = (_Float16)f.w;
    *(v4h*)(dst + (size_t)i * 4) = h;
  }
}

// ---------------------------------------------------------------------------
// Kernel 1: fused QKV projection, f16 WMMA with double-buffered async-to-LDS
// staging. grid = (M/128, N/64, 3), block = 256 (8 waves).
// Q,K stored [B,NH,S,HD]; V stored transposed [B,NH,HD,S] for the PV GEMM.
// ---------------------------------------------------------------------------
constexpr int LDA = 40;  // halfs; 80B row stride -> conflict-free b128 reads
constexpr int LDB = 40;

__device__ __forceinline__ void stage_tiles(const _Float16* __restrict__ xg,
                                            const _Float16* __restrict__ wg,
                                            _Float16* As, _Float16* Bs,
                                            int tid) {
  // A tile: 128 rows x 32 halfs (8KB) = 512 x 16B chunks -> 2 per thread
#pragma unroll
  for (int i = 0; i < 2; ++i) {
    const int id  = tid + i * 256;
    const int row = id >> 2;
    const int seg = id & 3;
#if USE_ASYNC_LDS
    __builtin_amdgcn_global_load_async_to_lds_b128(
        GPTR(xg + (size_t)row * Hv + seg * 8), LPTR(As + row * LDA + seg * 8),
        0, 0);
#else
    *(v8h*)(As + row * LDA + seg * 8) =
        *(const v8h*)(xg + (size_t)row * Hv + seg * 8);
#endif
  }
  // B tile: 64 rows x 32 halfs (4KB) = 256 x 16B chunks -> 1 per thread
  {
    const int row = tid >> 2;
    const int seg = tid & 3;
#if USE_ASYNC_LDS
    __builtin_amdgcn_global_load_async_to_lds_b128(
        GPTR(wg + (size_t)row * Hv + seg * 8), LPTR(Bs + row * LDB + seg * 8),
        0, 0);
#else
    *(v8h*)(Bs + row * LDB + seg * 8) =
        *(const v8h*)(wg + (size_t)row * Hv + seg * 8);
#endif
  }
}

__global__ __launch_bounds__(256) void bert_qkv_wmma(
    const _Float16* __restrict__ xh, const _Float16* __restrict__ wh,
    const float* __restrict__ bq, const float* __restrict__ bk,
    const float* __restrict__ bv, _Float16* __restrict__ qb,
    _Float16* __restrict__ kb, _Float16* __restrict__ vtb) {
  __shared__ _Float16 As[2][128 * LDA];
  __shared__ _Float16 Bs[2][64 * LDB];

  const int tid  = threadIdx.x;
  const int wave = tid >> 5;
  const int lane = tid & 31;
  const int half = lane >> 4;
  const int col  = lane & 15;
  const int m0   = blockIdx.x * 128;
  const int n0   = blockIdx.y * 64;
  const int mz   = blockIdx.z;  // 0=Q 1=K 2=V

  const _Float16* __restrict__ xg = xh + (size_t)m0 * Hv;
  const _Float16* __restrict__ wg =
      wh + (size_t)mz * Hv * Hv + (size_t)n0 * Hv;
  const float* __restrict__ bias = (mz == 0) ? bq : (mz == 1) ? bk : bv;

  v8f c[4] = {};

  stage_tiles(xg, wg, As[0], Bs[0], tid);  // prime k0 = 0

  for (int s = 0; s < 24; ++s) {
    const int cur = s & 1;
    // Issue next tile's async copies before waiting on the current one
    if (s + 1 < 24)
      stage_tiles(xg + (s + 1) * 32, wg + (s + 1) * 32, As[cur ^ 1],
                  Bs[cur ^ 1], tid);
#if USE_ASYNC_LDS
    if (s + 1 < 24)
      __builtin_amdgcn_s_wait_asynccnt(3);  // 3 in flight for next buffer
    else
      __builtin_amdgcn_s_wait_asynccnt(0);
#endif
    __syncthreads();  // all waves' copies for `cur` are visible

    const _Float16* arow = &As[cur][(wave * 16 + col) * LDA];
    const v16h a = ldA(arow + half * 16, arow + half * 16 + 8);
#pragma unroll
    for (int j = 0; j < 4; ++j) {
      const _Float16* brow = &Bs[cur][(j * 16 + col) * LDB + half * 16];
      const v16h b = ldA(brow, brow + 8);
      c[j] = wmma_f16(a, b, c[j]);
    }
    __syncthreads();  // everyone done reading `cur` before it is re-staged
  }

  // Epilogue: add bias, convert, scatter to head-major layouts
#pragma unroll
  for (int j = 0; j < 4; ++j) {
    const int n = n0 + j * 16 + col;
    const int h = n >> 6;
    const int d = n & 63;
    const float bval = bias[n];
#pragma unroll
    for (int r = 0; r < 8; ++r) {
      const int m    = m0 + wave * 16 + r + 8 * half;
      const int bidx = m >> 11;   // / Sv
      const int sm   = m & 2047;  // % Sv
      const _Float16 val = (_Float16)(c[j][r] + bval);
      if (mz == 2)
        vtb[(((size_t)bidx * NHv + h) * HDv + d) * Sv + sm] = val;
      else if (mz == 0)
        qb[(((size_t)bidx * NHv + h) * Sv + sm) * HDv + d] = val;
      else
        kb[(((size_t)bidx * NHv + h) * Sv + sm) * HDv + d] = val;
    }
  }
}

// ---------------------------------------------------------------------------
// Kernel 2: flash attention. grid = (S/64, B*NH), block = 128 (4 waves).
// Each wave owns 16 query rows; KV tiles of 64; online softmax; PV via WMMA
// with P transposed through LDS (C-layout f32 -> A-layout f16).
// ---------------------------------------------------------------------------
__global__ __launch_bounds__(128) void bert_attn_wmma(
    const _Float16* __restrict__ qb, const _Float16* __restrict__ kb,
    const _Float16* __restrict__ vtb, const float* __restrict__ mask,
    float* __restrict__ out) {
  constexpr int LDP = 72;  // halfs; 144B stride, 16B aligned chunks
  __shared__ _Float16 Ps[4 * 16 * LDP];

  const int tid  = threadIdx.x;
  const int wave = tid >> 5;
  const int lane = tid & 31;
  const int half = lane >> 4;
  const int col  = lane & 15;
  const int bh   = blockIdx.y;
  const int b    = bh / NHv;
  const int h    = bh % NHv;
  const int q0   = blockIdx.x * 64 + wave * 16;

  const _Float16* __restrict__ kbase = kb + (size_t)bh * Sv * HDv;
  const _Float16* __restrict__ vbase = vtb + (size_t)bh * HDv * Sv;
  const float* __restrict__ mrow     = mask + (size_t)b * Sv;

  // Q A-operands (d 0..31 and 32..63), loaded once
  const _Float16* qrow = qb + ((size_t)bh * Sv + q0 + col) * HDv;
  const v16h aq0 = ldA(qrow + half * 16, qrow + half * 16 + 8);
  const v16h aq1 = ldA(qrow + 32 + half * 16, qrow + 40 + half * 16);

  float mprev[8], lsum[8];
  v8f acc[4] = {};
#pragma unroll
  for (int r = 0; r < 8; ++r) { mprev[r] = -1e30f; lsum[r] = 0.0f; }

  _Float16* pw = &Ps[wave * 16 * LDP];

  for (int kv0 = 0; kv0 < Sv; kv0 += 64) {
    // ---- scores S = Q K^T for 16x64 tile (4 n-tiles x 2 K-steps) ----
    v8f s[4] = {};
#pragma unroll
    for (int j = 0; j < 4; ++j) {
      const _Float16* krow = kbase + (size_t)(kv0 + j * 16 + col) * HDv;
      const v16h b0 = ldA(krow + half * 16, krow + half * 16 + 8);
      const v16h b1 = ldA(krow + 32 + half * 16, krow + 40 + half * 16);
      s[j] = wmma_f16(aq0, b0, s[j]);
      s[j] = wmma_f16(aq1, b1, s[j]);
    }
    if (kv0 + 64 < Sv)  // uniform branch: prefetch next K tile into GL2
      __builtin_prefetch(kbase + (size_t)(kv0 + 64 + col + half * 16) * HDv, 0, 1);

    // ---- scale + additive mask + row max (16-wide shuffle reduce) ----
    float rmax[8];
#pragma unroll
    for (int j = 0; j < 4; ++j) {
      const float mval = mrow[kv0 + j * 16 + col];
#pragma unroll
      for (int r = 0; r < 8; ++r) {
        const float v = s[j][r] * 0.125f + mval;  // 1/sqrt(64)
        s[j][r] = v;
        rmax[r] = (j == 0) ? v : fmaxf(rmax[r], v);
      }
    }
#pragma unroll
    for (int r = 0; r < 8; ++r) {
      rmax[r] = fmaxf(rmax[r], __shfl_xor(rmax[r], 1, 32));
      rmax[r] = fmaxf(rmax[r], __shfl_xor(rmax[r], 2, 32));
      rmax[r] = fmaxf(rmax[r], __shfl_xor(rmax[r], 4, 32));
      rmax[r] = fmaxf(rmax[r], __shfl_xor(rmax[r], 8, 32));
    }

    // ---- online softmax update ----
    float alpha[8], rs[8];
#pragma unroll
    for (int r = 0; r < 8; ++r) {
      const float nm = fmaxf(mprev[r], rmax[r]);
      alpha[r] = __expf(mprev[r] - nm);
      mprev[r] = nm;
      rs[r] = 0.0f;
    }
#pragma unroll
    for (int j = 0; j < 4; ++j)
#pragma unroll
      for (int r = 0; r < 8; ++r) {
        const float p = __expf(s[j][r] - mprev[r]);
        s[j][r] = p;
        rs[r] += p;
      }
#pragma unroll
    for (int r = 0; r < 8; ++r) {
      rs[r] += __shfl_xor(rs[r], 1, 32);
      rs[r] += __shfl_xor(rs[r], 2, 32);
      rs[r] += __shfl_xor(rs[r], 4, 32);
      rs[r] += __shfl_xor(rs[r], 8, 32);
      lsum[r] = lsum[r] * alpha[r] + rs[r];
    }
#pragma unroll
    for (int jd = 0; jd < 4; ++jd)
#pragma unroll
      for (int r = 0; r < 8; ++r) acc[jd][r] *= alpha[r];

    // ---- stage P (C-layout -> LDS row-major f16), wave-private region ----
#pragma unroll
    for (int j = 0; j < 4; ++j)
#pragma unroll
      for (int r = 0; r < 8; ++r)
        pw[(r + 8 * half) * LDP + j * 16 + col] = (_Float16)s[j][r];
    __syncthreads();  // order DS stores before transposed reload

    // ---- reload P as A-operand, PV WMMA against V^T ----
    const _Float16* prow = &pw[col * LDP];
    const v16h p0 = ldA(prow + half * 16, prow + half * 16 + 8);
    const v16h p1 = ldA(prow + 32 + half * 16, prow + 40 + half * 16);
#pragma unroll
    for (int jd = 0; jd < 4; ++jd) {
      const _Float16* vrow = vbase + (size_t)(jd * 16 + col) * Sv + kv0;
      const v16h vb0 = ldA(vrow + half * 16, vrow + half * 16 + 8);
      const v16h vb1 = ldA(vrow + 32 + half * 16, vrow + 40 + half * 16);
      acc[jd] = wmma_f16(p0, vb0, acc[jd]);
      acc[jd] = wmma_f16(p1, vb1, acc[jd]);
    }
  }

  // ---- normalize by row sum, write [B,S,H] fp32 ----
#pragma unroll
  for (int r = 0; r < 8; ++r) {
    const float inv = 1.0f / lsum[r];
    const int srow  = q0 + r + 8 * half;
    float* orow = out + ((size_t)b * Sv + srow) * Hv + h * HDv;
#pragma unroll
    for (int jd = 0; jd < 4; ++jd) orow[jd * 16 + col] = acc[jd][r] * inv;
  }
}

// ---------------------------------------------------------------------------
extern "C" void kernel_launch(void* const* d_in, const int* in_sizes, int n_in,
                              void* d_out, int out_size, void* d_ws,
                              size_t ws_size, hipStream_t stream) {
  const float* x    = (const float*)d_in[0];
  const float* mask = (const float*)d_in[1];
  const float* Wq   = (const float*)d_in[2];
  const float* bq   = (const float*)d_in[3];
  const float* Wk   = (const float*)d_in[4];
  const float* bk   = (const float*)d_in[5];
  const float* Wv   = (const float*)d_in[6];
  const float* bv   = (const float*)d_in[7];

  // Workspace layout (f16): xh | wh[3] | q | k | v^T   (~104 MB total)
  const size_t nx  = (size_t)Bv * Sv * Hv;        // 12.58M
  const size_t nw  = (size_t)Hv * Hv;             // 589824
  const size_t per = (size_t)Bv * NHv * Sv * HDv; // 12.58M
  _Float16* xh  = (_Float16*)d_ws;
  _Float16* wh  = xh + nx;
  _Float16* qb  = wh + 3 * nw;
  _Float16* kb  = qb + per;
  _Float16* vtb = kb + per;

  // Phase 0: convert inputs to f16 once
  cvt_f32_to_f16<<<(int)(nx / 4 / 256), 256, 0, stream>>>(x, xh, (int)(nx / 4));
  cvt_f32_to_f16<<<(int)(nw / 4 / 256), 256, 0, stream>>>(Wq, wh, (int)(nw / 4));
  cvt_f32_to_f16<<<(int)(nw / 4 / 256), 256, 0, stream>>>(Wk, wh + nw, (int)(nw / 4));
  cvt_f32_to_f16<<<(int)(nw / 4 / 256), 256, 0, stream>>>(Wv, wh + 2 * nw, (int)(nw / 4));

  // Phase 1: QKV projections (async double-buffered WMMA GEMM)
  dim3 g1(16384 / 128, Hv / 64, 3);  // (128, 12, 3)
  bert_qkv_wmma<<<g1, 256, 0, stream>>>(xh, wh, bq, bk, bv, qb, kb, vtb);

  // Phase 2: flash attention
  dim3 g2(Sv / 64, Bv * NHv);  // (32, 96)
  bert_attn_wmma<<<g2, 128, 0, stream>>>(qb, kb, vtb, mask, (float*)d_out);
}
